// GRU_unit_77867757076553
// MI455X (gfx1250) — matrix operand order repaired and checked by reading, compiled, not run on hardware
//
#include <hip/hip_runtime.h>

// ---------------------------------------------------------------------------
// GRU cell on MI455X (gfx1250): bf16 WMMA GEMMs (f32 accumulate), f32 gates.
// B=16384, E=H=1024. Triple-buffered async global->LDS pipeline (ASYNCcnt),
// one barrier per K-step, addrspace(3) fragment reads -> ds_load_b128.
// ---------------------------------------------------------------------------

#define GRU_B 16384
#define GRU_K 1024   // E == H == 1024
#define LDA   40     // padded LDS row stride in bf16 elements (80B: bank-conflict free)

#define AS3 __attribute__((address_space(3)))

typedef __attribute__((ext_vector_type(16))) __bf16        v16bf;
typedef __attribute__((ext_vector_type(8)))  float         v8f;
typedef __attribute__((ext_vector_type(4)))  unsigned int  u32x4;

union ABu {
    v16bf v;
    u32x4 q[2];
};

__device__ __forceinline__ unsigned short f32_to_bf16(float f) {
    unsigned int u = __float_as_uint(f);
    unsigned int r = (u + 0x7FFFu + ((u >> 16) & 1u)) >> 16;
    return (unsigned short)r;
}

__device__ __forceinline__ float fast_sigmoid(float x) {
    return 1.0f / (1.0f + __expf(-x));
}

__device__ __forceinline__ float fast_tanh(float x) {
    x = fminf(fmaxf(x, -15.0f), 15.0f);
    float e = __expf(-2.0f * x);
    return (1.0f - e) / (1.0f + e);
}

// --------------------------- f32 -> bf16 convert ---------------------------
__global__ __launch_bounds__(256) void cvt_bf16_kernel(
    const float* __restrict__ src, unsigned short* __restrict__ dst, int n) {
    int i = (blockIdx.x * 256 + threadIdx.x) * 4;
    if (i + 3 < n) {
        float4 f = *(const float4*)(src + i);
        uint2 o;
        o.x = (unsigned int)f32_to_bf16(f.x) | ((unsigned int)f32_to_bf16(f.y) << 16);
        o.y = (unsigned int)f32_to_bf16(f.z) | ((unsigned int)f32_to_bf16(f.w) << 16);
        *(uint2*)(dst + i) = o;
    }
}

// ------------------- weight transpose + convert to bf16 --------------------
struct WPtrs {
    const float*    w[6];
    unsigned short* t[6];
};

__global__ __launch_bounds__(256) void transpose_cvt_kernel(WPtrs p) {
    __shared__ float tile[32][33];
    const float*    W = p.w[blockIdx.z];
    unsigned short* T = p.t[blockIdx.z];
    int tx = threadIdx.x;          // 0..31
    int ty = threadIdx.y;          // 0..7
    int n0 = blockIdx.x * 32;
    int k0 = blockIdx.y * 32;
#pragma unroll
    for (int j = 0; j < 32; j += 8)
        tile[ty + j][tx] = W[(size_t)(k0 + ty + j) * GRU_K + (n0 + tx)];
    __syncthreads();
#pragma unroll
    for (int j = 0; j < 32; j += 8)
        T[(size_t)(n0 + ty + j) * GRU_K + (k0 + tx)] = f32_to_bf16(tile[tx][ty + j]);
}

// ------------------------------ WMMA helper --------------------------------
__device__ __forceinline__ v8f wmma_bf16(const v16bf& a, const v16bf& b, v8f c) {
    return __builtin_amdgcn_wmma_f32_16x16x32_bf16(
        /*neg_a=*/false, a, /*neg_b=*/false, b,
        /*c_mod=*/(short)0, c, /*reuse_a=*/false, /*reuse_b=*/false);
}

// ---------------------- async global -> LDS (16 bytes) ---------------------
__device__ __forceinline__ void async_load_b128(AS3 unsigned short* lds_ptr,
                                                const void* gaddr) {
    unsigned int lds_addr = (unsigned int)(uintptr_t)lds_ptr;  // AS3 ptr == LDS byte addr
    asm volatile("global_load_async_to_lds_b128 %0, %1, off"
                 :: "v"(lds_addr), "v"((unsigned long long)(uintptr_t)gaddr)
                 : "memory");
}

// Issue one 128x32 A tile + 128x32 B tile (16 KB) as 4 async b128 per thread.
__device__ __forceinline__ void issue_tile(
    const unsigned short* __restrict__ gA, const unsigned short* __restrict__ gWt,
    AS3 unsigned short* sAb, AS3 unsigned short* sBb, int m0, int n0, int k0, int t) {
    const int r  = t >> 2;       // 0..63
    const int ce = (t & 3) * 8;  // 16B chunk within the 32-element K-slab
    async_load_b128(sAb + r * LDA + ce,
                    gA + (size_t)(m0 + r) * GRU_K + k0 + ce);
    async_load_b128(sAb + (r + 64) * LDA + ce,
                    gA + (size_t)(m0 + r + 64) * GRU_K + k0 + ce);
    async_load_b128(sBb + r * LDA + ce,
                    gWt + (size_t)(n0 + r) * GRU_K + k0 + ce);
    async_load_b128(sBb + (r + 64) * LDA + ce,
                    gWt + (size_t)(n0 + r + 64) * GRU_K + k0 + ce);
}

// 8 WMMAs of one K=32 step from LDS (wave tile 64M x 32N).
__device__ __forceinline__ void compute_step(
    const AS3 unsigned short* sAb, const AS3 unsigned short* sBb,
    int aRow0, int bRow0, int half, v8f acc[4][2]) {
    ABu fb[2];
#pragma unroll
    for (int nt = 0; nt < 2; ++nt) {
        const AS3 unsigned short* p = sBb + (bRow0 + nt * 16) * LDA + half * 16;
        fb[nt].q[0] = *(const AS3 u32x4*)(p);
        fb[nt].q[1] = *(const AS3 u32x4*)(p + 8);
    }
#pragma unroll
    for (int mt = 0; mt < 4; ++mt) {
        ABu fa;
        const AS3 unsigned short* p = sAb + (aRow0 + mt * 16) * LDA;
        fa.q[0] = *(const AS3 u32x4*)(p + half * 16);
        fa.q[1] = *(const AS3 u32x4*)(p + half * 16 + 8);
        acc[mt][0] = wmma_bf16(fa.v, fb[0].v, acc[mt][0]);
        acc[mt][1] = wmma_bf16(fa.v, fb[1].v, acc[mt][1]);
    }
}

// Full K=1024 GEMM accumulate: triple-buffered async LDS pipeline,
// issue 2 tiles ahead, ONE barrier per K-step (wait precedes barrier, so the
// barrier both publishes buffer s and retires all readers of buffer s-1
// before buffer (s+2)%3 == (s-1)%3 is re-issued).
__device__ __forceinline__ void gemm_lds(
    const unsigned short* __restrict__ gA, const unsigned short* __restrict__ gWt,
    AS3 unsigned short* A0, AS3 unsigned short* A1, AS3 unsigned short* A2,
    AS3 unsigned short* B0, AS3 unsigned short* B1, AS3 unsigned short* B2,
    int m0, int n0, int t, int aRow0, int bRow0, int half, v8f acc[4][2]) {
    const int NSTEP = GRU_K / 32;  // 32
    __syncthreads();               // prior readers of LDS are done
    issue_tile(gA, gWt, A0, B0, m0, n0, 0, t);
    issue_tile(gA, gWt, A1, B1, m0, n0, 32, t);
    AS3 unsigned short* cA = A0; AS3 unsigned short* nA = A1; AS3 unsigned short* pA = A2;
    AS3 unsigned short* cB = B0; AS3 unsigned short* nB = B1; AS3 unsigned short* pB = B2;
    for (int s = 0; s < NSTEP; ++s) {
        if (s + 1 < NSTEP) {
            asm volatile("s_wait_asynccnt 4" ::: "memory");   // buffer s landed
        } else {
            asm volatile("s_wait_asynccnt 0" ::: "memory");
        }
        __syncthreads();           // publish buffer s; retire readers of s-1
        if (s + 2 < NSTEP)
            issue_tile(gA, gWt, pA, pB, m0, n0, (s + 2) * 32, t);
        compute_step(cA, cB, aRow0, bRow0, half, acc);
        AS3 unsigned short* tA = cA; cA = nA; nA = pA; pA = tA;
        AS3 unsigned short* tB = cB; cB = nB; nB = pB; pB = tB;
    }
}

// --------------------------- fused dual-GEMM pass --------------------------
//   MODE 0 (r gate)   : outb = bf16( sigmoid(acc+b) * hf )          [hr]
//   MODE 1 (z gate)   : outf = sigmoid(acc+b)                       [zt]
//   MODE 2 (candidate): outf = (1-zt)*hf + zt*tanh(acc+b)           [h_new]
template <int MODE>
__global__ __launch_bounds__(256) void gru_gemm2_kernel(
    const unsigned short* __restrict__ A1, const unsigned short* __restrict__ W1t,
    const unsigned short* __restrict__ A2, const unsigned short* __restrict__ W2t,
    const float* __restrict__ bias,
    const float* __restrict__ hf,
    const float* __restrict__ ztf,
    float* __restrict__ outf,
    unsigned short* __restrict__ outb) {
    __shared__ unsigned short sA[3][128 * LDA];
    __shared__ unsigned short sB[3][128 * LDA];

    AS3 unsigned short* sA0 = (AS3 unsigned short*)&sA[0][0];
    AS3 unsigned short* sA1 = (AS3 unsigned short*)&sA[1][0];
    AS3 unsigned short* sA2 = (AS3 unsigned short*)&sA[2][0];
    AS3 unsigned short* sB0 = (AS3 unsigned short*)&sB[0][0];
    AS3 unsigned short* sB1 = (AS3 unsigned short*)&sB[1][0];
    AS3 unsigned short* sB2 = (AS3 unsigned short*)&sB[2][0];

    const int t    = threadIdx.x;
    const int lane = t & 31;
    const int half = lane >> 4;
    const int l16  = lane & 15;
    const int wave = t >> 5;             // 0..7
    const int wm   = wave & 1;           // 2 waves along M (64 rows each)
    const int wn   = wave >> 1;          // 4 waves along N (32 cols each)

    const int m0 = blockIdx.x * 128;     // macro-tile origin
    const int n0 = blockIdx.y * 128;
    const int aRow0 = wm * 64 + l16;     // LDS-local A fragment row (mt adds 16)
    const int bRow0 = wn * 32 + l16;     // LDS-local B fragment row (nt adds 16)

    v8f acc[4][2] = {};

    gemm_lds(A1, W1t, sA0, sA1, sA2, sB0, sB1, sB2,
             m0, n0, t, aRow0, bRow0, half, acc);
    gemm_lds(A2, W2t, sA0, sA1, sA2, sB0, sB1, sB2,
             m0, n0, t, aRow0, bRow0, half, acc);

#pragma unroll
    for (int nt = 0; nt < 2; ++nt) {
        const int col = n0 + wn * 32 + nt * 16 + l16;
        const float bv = bias[col];
#pragma unroll
        for (int mt = 0; mt < 4; ++mt) {
#pragma unroll
            for (int v = 0; v < 8; ++v) {
                const int row = m0 + wm * 64 + mt * 16 + half * 8 + v;
                const size_t idx = (size_t)row * GRU_K + col;
                const float pre = acc[mt][nt][v] + bv;
                if (MODE == 0) {
                    float r = fast_sigmoid(pre);
                    outb[idx] = f32_to_bf16(r * hf[idx]);
                } else if (MODE == 1) {
                    outf[idx] = fast_sigmoid(pre);
                } else {
                    float z  = ztf[idx];
                    float ht = fast_tanh(pre);
                    outf[idx] = (1.0f - z) * hf[idx] + z * ht;
                }
            }
        }
    }
}

// ------------------------------- launcher ----------------------------------
extern "C" void kernel_launch(void* const* d_in, const int* in_sizes, int n_in,
                              void* d_out, int out_size, void* d_ws, size_t ws_size,
                              hipStream_t stream) {
    (void)in_sizes; (void)n_in; (void)out_size; (void)ws_size;

    const float* x    = (const float*)d_in[0];
    const float* h    = (const float*)d_in[1];
    const float* W_ri = (const float*)d_in[2];
    const float* b_ri = (const float*)d_in[3];
    const float* W_rh = (const float*)d_in[4];
    const float* W_zi = (const float*)d_in[5];
    const float* b_zi = (const float*)d_in[6];
    const float* W_zh = (const float*)d_in[7];
    const float* W_hi = (const float*)d_in[8];
    const float* b_hi = (const float*)d_in[9];
    const float* W_hh = (const float*)d_in[10];

    // ---- workspace carve-up (256B aligned) ----
    size_t off = 0;
    char* base = (char*)d_ws;
    auto carve = [&](size_t bytes) -> void* {
        void* p = base + off;
        off += (bytes + 255) & ~(size_t)255;
        return p;
    };
    const size_t BH = (size_t)GRU_B * GRU_K;
    unsigned short* xb  = (unsigned short*)carve(BH * 2);
    unsigned short* hb  = (unsigned short*)carve(BH * 2);
    unsigned short* hrb = (unsigned short*)carve(BH * 2);
    float*          ztf = (float*)carve(BH * 4);
    unsigned short* wt[6];
    for (int i = 0; i < 6; ++i)
        wt[i] = (unsigned short*)carve((size_t)GRU_K * GRU_K * 2);
    // wt order: ri, rh, zi, zh, hi, hh

    // ---- 1) convert x, h to bf16 ----
    {
        int n = (int)BH;
        int blocks = n / (256 * 4);
        cvt_bf16_kernel<<<blocks, 256, 0, stream>>>(x, xb, n);
        cvt_bf16_kernel<<<blocks, 256, 0, stream>>>(h, hb, n);
    }

    // ---- 2) transpose+convert all six weights ----
    {
        WPtrs p;
        p.w[0] = W_ri; p.w[1] = W_rh; p.w[2] = W_zi;
        p.w[3] = W_zh; p.w[4] = W_hi; p.w[5] = W_hh;
        for (int i = 0; i < 6; ++i) p.t[i] = wt[i];
        dim3 g(GRU_K / 32, GRU_K / 32, 6);
        transpose_cvt_kernel<<<g, dim3(32, 8), 0, stream>>>(p);
    }

    // ---- 3) three fused dual-GEMM passes (128x128 macro-tiles) ----
    dim3 gg(GRU_B / 128, GRU_K / 128);
    // r gate -> hr (bf16)
    gru_gemm2_kernel<0><<<gg, 256, 0, stream>>>(
        xb, wt[0], hb, wt[1], b_ri, h, nullptr, nullptr, hrb);
    // z gate -> zt (f32)
    gru_gemm2_kernel<1><<<gg, 256, 0, stream>>>(
        xb, wt[2], hb, wt[3], b_zi, nullptr, nullptr, ztf, nullptr);
    // candidate + blend -> d_out (f32)
    gru_gemm2_kernel<2><<<gg, 256, 0, stream>>>(
        xb, wt[4], hrb, wt[5], b_hi, h, ztf, (float*)d_out, nullptr);
}
